// QuantMlp_84920093376702
// MI455X (gfx1250) — matrix operand order repaired and checked by reading, not compiled, not run on hardware
//
#include <hip/hip_runtime.h>
#include <hip/hip_bf16.h>

// ---------------------------------------------------------------------------
// Types for CDNA5 WMMA
// ---------------------------------------------------------------------------
typedef __attribute__((ext_vector_type(16))) __bf16       v16bf;
typedef __attribute__((ext_vector_type(8)))  float        v8f;
typedef __attribute__((ext_vector_type(4)))  unsigned int u32x4;

struct FragU { u32x4 lo; u32x4 hi; };   // 32 bytes == 16 bf16

#define WMMA_BF16(A_, B_, C_) \
  __builtin_amdgcn_wmma_f32_16x16x32_bf16(false, (A_), false, (B_), (short)0, (C_), false, false)

__device__ __forceinline__ unsigned short f2bf(float f) {
  unsigned int u = __float_as_uint(f);
  u += 0x7FFFu + ((u >> 16) & 1u);        // round-to-nearest-even
  return (unsigned short)(u >> 16);
}
__device__ __forceinline__ float bf2f(unsigned short h) {
  return __uint_as_float(((unsigned int)h) << 16);
}
__device__ __forceinline__ float gelu_exact(float x) {
  return 0.5f * x * (1.0f + erff(x * 0.70710678118654752440f));
}

// CDNA5 async global->LDS (ASYNCcnt path). Inline asm: portable across
// ROCm7.2 / amdgpu-toolchain (builtin arity differs between them).
__device__ __forceinline__ void async_ld_b128(unsigned ldsAddr, const void* g) {
  asm volatile("global_load_async_to_lds_b128 %0, %1, off"
               :: "v"(ldsAddr), "v"((unsigned long long)(__SIZE_TYPE__)g)
               : "memory");
}
__device__ __forceinline__ void wait_async0() {
  asm volatile("s_wait_asynccnt 0" ::: "memory");
}

#define BM 128
#define BN 128
#define BK 32
#define LDSS 40      // padded bf16 row stride (80B, 16B aligned)

// ---------------------------------------------------------------------------
// Weight conversion fp32 -> bf16 (W1 and W2 have identical element counts)
// ---------------------------------------------------------------------------
__global__ void cvt_weights_kernel(const float* __restrict__ W1, const float* __restrict__ W2,
                                   unsigned short* __restrict__ W1b, unsigned short* __restrict__ W2b,
                                   long long n) {
  long long i = (long long)blockIdx.x * blockDim.x + threadIdx.x;
  if (i < n) { W1b[i] = f2bf(W1[i]); W2b[i] = f2bf(W2[i]); }
}

// ---------------------------------------------------------------------------
// Fused channel-gather + per-token dynamic int8 fake-quant of x.
// absmax is permutation-invariant -> pass 1 reads the row coalesced.
// Output Qx holds integer-valued bf16 (exact), sx holds the per-token scale.
// ---------------------------------------------------------------------------
__global__ __launch_bounds__(256) void quant_x_kernel(const float* __restrict__ x,
                                                      const int* __restrict__ ridx,
                                                      unsigned short* __restrict__ Qx,
                                                      float* __restrict__ sx, int D) {
  const int token = blockIdx.x;
  const float* row = x + (size_t)token * D;
  __shared__ float red[8];

  float lmax = 0.f;
  for (int d = threadIdx.x; d < D; d += blockDim.x) lmax = fmaxf(lmax, fabsf(row[d]));
  #pragma unroll
  for (int off = 16; off > 0; off >>= 1) lmax = fmaxf(lmax, __shfl_xor(lmax, off, 32));
  if ((threadIdx.x & 31) == 0) red[threadIdx.x >> 5] = lmax;
  __syncthreads();
  if (threadIdx.x == 0) {
    float m = red[0];
    #pragma unroll
    for (int w = 1; w < 8; ++w) m = fmaxf(m, red[w]);
    float s = fmaxf(m * (1.0f / 127.0f), 1e-8f);
    sx[token] = s;
    red[0] = 1.0f / s;
  }
  __syncthreads();
  const float rs = red[0];
  unsigned short* qrow = Qx + (size_t)token * D;
  for (int d = threadIdx.x; d < D; d += blockDim.x) {
    float v = row[ridx[d]];
    float q = fminf(fmaxf(rintf(v * rs), -128.0f), 127.0f);
    qrow[d] = f2bf(q);   // |q| <= 128 -> exact in bf16
  }
}

__global__ void zero_u32_kernel(unsigned int* __restrict__ p, int n) {
  int i = blockIdx.x * blockDim.x + threadIdx.x;
  if (i < n) p[i] = 0u;
}

__global__ void scales_kernel(const unsigned int* __restrict__ amax,
                              float* __restrict__ sh, float* __restrict__ rsh, int M) {
  int i = blockIdx.x * blockDim.x + threadIdx.x;
  if (i < M) {
    float a = __uint_as_float(amax[i]);
    float s = fmaxf(a * (1.0f / 127.0f), 1e-8f);
    sh[i] = s;
    rsh[i] = 1.0f / s;
  }
}

// In-place requant of GELU output to integer-valued bf16.
__global__ void quant_h_kernel(unsigned short* __restrict__ Hbuf,
                               const float* __restrict__ rsh, int Hdim, long long total) {
  long long i = (long long)blockIdx.x * blockDim.x + threadIdx.x;
  if (i < total) {
    int m = (int)(i / Hdim);
    float h = bf2f(Hbuf[i]);
    float q = fminf(fmaxf(rintf(h * rsh[m]), -128.0f), 127.0f);
    Hbuf[i] = f2bf(q);
  }
}

// ---------------------------------------------------------------------------
// Tiled WMMA GEMM:  C(M,N) = A(M,K) @ Bt(N,K)^T, deferred per-row scale.
// Block = 128x128 tile, 256 threads = 8 waves (4 along M x 2 along N).
// Each wave: 32x64 = 2x4 grid of 16x16x32 bf16 WMMA tiles.
// Tile staging: async global->LDS (ASYNCcnt) into a double-buffered LDS
// pair; one barrier per K-tile; fetch of tile k+1 overlaps WMMA on tile k.
// EPI==1: h = GELU(acc*sx[m] + b1[n]) -> bf16 + per-row absmax atomics.
// EPI==2: out = acc*sh[m] + b2[n] -> fp32.
// ---------------------------------------------------------------------------
template <int EPI>
__global__ __launch_bounds__(256) void wmma_gemm_kernel(
    const unsigned short* __restrict__ A,    // M x K bf16 (integer-valued)
    const unsigned short* __restrict__ Bt,   // N x K bf16 (weight rows = out channels)
    const float* __restrict__ rowScale,      // M
    const float* __restrict__ bias,          // N
    unsigned short* __restrict__ outH,       // EPI==1
    unsigned int*  __restrict__ amax,        // EPI==1
    float* __restrict__ outF,                // EPI==2
    int M, int N, int K) {
  (void)M;
  __shared__ __align__(16) unsigned short As[2][BM * LDSS];
  __shared__ __align__(16) unsigned short Bs[2][BN * LDSS];

  const int tid  = threadIdx.x;
  const int wave = tid >> 5;
  const int lane = tid & 31;
  const int laneCol  = lane & 15;
  const int laneHalf = lane >> 4;
  const int wm = wave & 3;    // 4 waves along M: 32 rows each
  const int wn = wave >> 2;   // 2 waves along N: 64 cols each

  const int m0 = blockIdx.y * BM;
  const int n0 = blockIdx.x * BN;

  // cooperative tile loads: thread -> (row, 16-elem segment)
  const int lrow = tid >> 1;
  const int lcol = (tid & 1) << 4;
  const unsigned short* gA = A  + (size_t)(m0 + lrow) * K + lcol;
  const unsigned short* gB = Bt + (size_t)(n0 + lrow) * K + lcol;
  const unsigned ldsA0 = (unsigned)(__SIZE_TYPE__)&As[0][lrow * LDSS + lcol];
  const unsigned ldsA1 = (unsigned)(__SIZE_TYPE__)&As[1][lrow * LDSS + lcol];
  const unsigned ldsB0 = (unsigned)(__SIZE_TYPE__)&Bs[0][lrow * LDSS + lcol];
  const unsigned ldsB1 = (unsigned)(__SIZE_TYPE__)&Bs[1][lrow * LDSS + lcol];

  v8f acc[2][4];
  #pragma unroll
  for (int mt = 0; mt < 2; ++mt)
    #pragma unroll
    for (int nt = 0; nt < 4; ++nt)
      acc[mt][nt] = (v8f){0.f, 0.f, 0.f, 0.f, 0.f, 0.f, 0.f, 0.f};

  // prologue: async-stage tile k0=0 into buffer 0
  async_ld_b128(ldsA0,      gA);
  async_ld_b128(ldsA0 + 16, gA + 8);
  async_ld_b128(ldsB0,      gB);
  async_ld_b128(ldsB0 + 16, gB + 8);

  int cur = 0;
  for (int k0 = 0; k0 < K; k0 += BK) {
    wait_async0();                          // my tile-k loads have landed in LDS
    __syncthreads();                        // everyone's landed; nobody reads buf cur^1
    const bool hasNext = (k0 + BK) < K;
    if (hasNext) {                          // overlap next async fetch with WMMA burst
      const unsigned la = cur ? ldsA0 : ldsA1;
      const unsigned lb = cur ? ldsB0 : ldsB1;
      async_ld_b128(la,      gA + k0 + BK);
      async_ld_b128(la + 16, gA + k0 + BK + 8);
      async_ld_b128(lb,      gB + k0 + BK);
      async_ld_b128(lb + 16, gB + k0 + BK + 8);
      if (k0 + 2 * BK < K) __builtin_prefetch(gB + k0 + 2 * BK, 0, 1);
    }

    // A fragments (ISA 16-bit 16x32 layout): slots 0-7 = K[half*8+j], 8-15 = K[16+half*8+j]
    v16bf af[2];
    #pragma unroll
    for (int mt = 0; mt < 2; ++mt) {
      const int mr = wm * 32 + mt * 16 + laneCol;
      const u32x4* pa = (const u32x4*)(&As[cur][mr * LDSS + laneHalf * 8]);
      FragU f; f.lo = pa[0]; f.hi = pa[2];
      af[mt] = __builtin_bit_cast(v16bf, f);
    }
    // B fragments (32x16): slots 0-7 = K[half*8+j], 8-15 = K[16+half*8+j], N = laneCol
    #pragma unroll
    for (int nt = 0; nt < 4; ++nt) {
      const int nr = wn * 64 + nt * 16 + laneCol;
      const u32x4* pb = (const u32x4*)(&Bs[cur][nr * LDSS + laneHalf * 8]);
      FragU f; f.lo = pb[0]; f.hi = pb[2];
      v16bf bf = __builtin_bit_cast(v16bf, f);
      acc[0][nt] = WMMA_BF16(af[0], bf, acc[0][nt]);
      acc[1][nt] = WMMA_BF16(af[1], bf, acc[1][nt]);
    }
    cur ^= 1;
  }

  // epilogue: C/D layout -> element (row = half*8 + r, col = laneCol) in VGPR r
  #pragma unroll
  for (int mt = 0; mt < 2; ++mt) {
    #pragma unroll
    for (int r = 0; r < 8; ++r) {
      const int gm = m0 + wm * 32 + mt * 16 + laneHalf * 8 + r;
      const float s = rowScale[gm];
      float lmax = 0.f;
      #pragma unroll
      for (int nt = 0; nt < 4; ++nt) {
        const int gn = n0 + wn * 64 + nt * 16 + laneCol;
        float v = acc[mt][nt][r] * s + bias[gn];
        if (EPI == 1) {
          float g = gelu_exact(v);
          outH[(size_t)gm * N + gn] = f2bf(g);
          lmax = fmaxf(lmax, fabsf(g));
        } else {
          outF[(size_t)gm * N + gn] = v;
        }
      }
      if (EPI == 1) {
        // lanes 0..15 (and 16..31) hold the same row set -> reduce over cols
        #pragma unroll
        for (int off = 1; off < 16; off <<= 1)
          lmax = fmaxf(lmax, __shfl_xor(lmax, off, 32));
        if (laneCol == 0) atomicMax(&amax[gm], __float_as_uint(lmax));
      }
    }
  }
}

// ---------------------------------------------------------------------------
// Host-side orchestration
// ---------------------------------------------------------------------------
extern "C" void kernel_launch(void* const* d_in, const int* in_sizes, int n_in,
                              void* d_out, int out_size, void* d_ws, size_t ws_size,
                              hipStream_t stream) {
  (void)n_in; (void)out_size; (void)ws_size;
  const float* x   = (const float*)d_in[0];
  const int*   rix = (const int*)  d_in[1];
  const float* W1  = (const float*)d_in[2];
  const float* b1  = (const float*)d_in[3];
  const float* W2  = (const float*)d_in[4];
  const float* b2  = (const float*)d_in[5];

  const int D = in_sizes[1];            // 1152
  const int H = in_sizes[3];            // 4608
  const int M = in_sizes[0] / D;        // 32768 tokens

  // workspace carve-out (256B aligned)
  char* ws = (char*)d_ws;
  size_t off = 0;
  auto carve = [&](size_t bytes) -> char* {
    char* p = ws + off;
    off = (off + bytes + 255) & ~(size_t)255;
    return p;
  };
  unsigned short* W1b = (unsigned short*)carve((size_t)H * D * 2);
  unsigned short* W2b = (unsigned short*)carve((size_t)H * D * 2);
  unsigned short* Qx  = (unsigned short*)carve((size_t)M * D * 2);
  unsigned short* Hq  = (unsigned short*)carve((size_t)M * H * 2);
  float*        sx    = (float*)        carve((size_t)M * 4);
  float*        sh    = (float*)        carve((size_t)M * 4);
  float*        rsh   = (float*)        carve((size_t)M * 4);
  unsigned int* amax  = (unsigned int*) carve((size_t)M * 4);

  const long long nW = (long long)H * D;
  cvt_weights_kernel<<<(unsigned)((nW + 255) / 256), 256, 0, stream>>>(W1, W2, W1b, W2b, nW);
  quant_x_kernel<<<M, 256, 0, stream>>>(x, rix, Qx, sx, D);
  zero_u32_kernel<<<(M + 255) / 256, 256, 0, stream>>>(amax, M);

  dim3 g1(H / BN, M / BM);
  wmma_gemm_kernel<1><<<g1, 256, 0, stream>>>(Qx, W1b, sx, b1, Hq, amax, nullptr, M, H, D);

  scales_kernel<<<(M + 255) / 256, 256, 0, stream>>>(amax, sh, rsh, M);
  const long long tot = (long long)M * H;
  quant_h_kernel<<<(unsigned)((tot + 255) / 256), 256, 0, stream>>>(Hq, rsh, H, tot);

  dim3 g2(D / BN, M / BM);
  wmma_gemm_kernel<2><<<g2, 256, 0, stream>>>(Hq, W2b, sh, b2, nullptr, nullptr, (float*)d_out, M, D, H);
}
